// SingleViewTransformer_14035953123837
// MI455X (gfx1250) — compile-verified, matrix operand order checked by reading, not run on hardware
//
#include <hip/hip_runtime.h>
#include <math.h>

// ---------------------------------------------------------------------------
// SingleViewTransformer for MI455X (gfx1250): wave32 + v_wmma_f32_16x16x32_f16
// B=8192 anchors, K=64 neighbors, D=64, H=4 heads (dh=16), FF=128, 2 layers.
// Weights pre-swizzled into WMMA B-fragment order (contiguous 32B per lane);
// GEMM epilogues stage C strips in LDS and flush with coalesced b128 stores.
// ---------------------------------------------------------------------------

typedef _Float16 h16;
typedef __attribute__((ext_vector_type(16))) _Float16 v16h;
typedef __attribute__((ext_vector_type(8)))  _Float16 v8h;
typedef __attribute__((ext_vector_type(8)))  float    v8f;
typedef __attribute__((ext_vector_type(4)))  float    v4f;

#define NBATCH 8192
#define KNEI   64

#define STAGE_LD 68   // padded row stride (floats) for the LDS stage tile

// ------------------------- WMMA fragment helpers ---------------------------

__device__ __forceinline__ v8f wmma16x16x32(v16h a, v16h b, v8f c) {
  return __builtin_amdgcn_wmma_f32_16x16x32_f16(false, a, false, b, (short)0, c,
                                                false, false);
}

// A fragment: 16x32 f16, row-major source (lda halfs per row).
__device__ __forceinline__ v16h load_a_frag_h(const h16* A, int lda, int kOff) {
  int lane = threadIdx.x & 31;
  int m = lane & 15;
  int kb = (lane >> 4) * 8;
  const v8h* p = (const v8h*)(A + (size_t)m * lda + kOff + kb);
  v8h lo = p[0];
  v8h hi = p[2];           // +16 halfs
  v16h r;
#pragma unroll
  for (int i = 0; i < 8; ++i) { r[i] = lo[i]; r[8 + i] = hi[i]; }
  return r;
}

// A fragment from f32 source (embedding GEMM, K-dim = 32).
__device__ __forceinline__ v16h load_a_frag_f(const float* A, int lda, int kOff) {
  int lane = threadIdx.x & 31;
  int m = lane & 15;
  int kb = (lane >> 4) * 8;
  const v8f* p = (const v8f*)(A + (size_t)m * lda + kOff + kb);
  v8f lo = p[0];
  v8f hi = p[2];
  v16h r;
#pragma unroll
  for (int i = 0; i < 8; ++i) { r[i] = (h16)lo[i]; r[8 + i] = (h16)hi[i]; }
  return r;
}

// B fragment from pre-swizzled weight buffer: one contiguous 32B load per lane.
__device__ __forceinline__ v16h load_b_swz(const h16* Wsw, int NT, int kt, int nt) {
  int lane = threadIdx.x & 31;
  return *(const v16h*)(Wsw + (size_t)(((kt * NT + nt) * 32) + lane) * 16);
}

__device__ __forceinline__ float gelu_f(float x) {
  const float k0 = 0.7978845608028654f; // sqrt(2/pi)
  return 0.5f * x * (1.0f + tanhf(k0 * (x + 0.044715f * x * x * x)));
}

// ------------------------------ pack kernels -------------------------------

__global__ void pack_f16_kernel(const float* __restrict__ src,
                                h16* __restrict__ dst, int n) {
  int i = blockIdx.x * blockDim.x + threadIdx.x;
  if (i < n) dst[i] = (h16)src[i];
}

// Swizzle W[KD,N] (row-major f32) into WMMA B-fragment order:
// dst[(((kt*NT+nt)*32)+lane)*16 + e] = W[k, n]  with the 16x16x32 layout.
__global__ void pack_swz_kernel(const float* __restrict__ W,
                                h16* __restrict__ dst, int KD, int N) {
  int total = KD * N;
  int NT = N >> 4;
  int i = blockIdx.x * blockDim.x + threadIdx.x;
  if (i >= total) return;
  int e  = i & 15;
  int l  = (i >> 4) & 31;
  int r  = i >> 9;
  int nt = r % NT;
  int kt = r / NT;
  int n  = nt * 16 + (l & 15);
  int kb = (l >> 4) * 8;
  int ko = (e < 8) ? (kb + e) : (kb + 16 + (e - 8));
  int k  = kt * 32 + ko;
  dst[i] = (h16)W[(size_t)k * N + n];
}

// pi one-hot at view 1, tau = exp(log_tau)
__global__ void fill_pi_tau_kernel(float* __restrict__ out,
                                   const float* __restrict__ log_tau) {
  int i = blockIdx.x * blockDim.x + threadIdx.x;
  float* pi = out + NBATCH * 8;
  if (i < NBATCH * 16) pi[i] = ((i & 3) == 1) ? 1.0f : 0.0f;
  if (i < 4) out[NBATCH * 8 + NBATCH * 16 + i] = __expf(log_tau[i]);
}

// -------------------- staged, coalesced epilogue flush ---------------------
// stage: 16 x STAGE_LD floats (one wave's 16x64 column group). Wave-local:
// ds writes then ds reads within the same wave; compiler orders via dscnt.
template <bool GELU, bool RES, bool OUTF16, bool OUTF32>
__device__ __forceinline__ void wave_flush(const float* __restrict__ stage,
                                           const float* __restrict__ resIn,
                                           h16* __restrict__ outH,
                                           float* __restrict__ outF,
                                           int row0, int colBase, int N) {
  int lane = threadIdx.x & 31;
  int rr = lane >> 1;
  int c0 = (lane & 1) * 32;
  const float* srow = stage + rr * STAGE_LD + c0;
  size_t gbase = (size_t)(row0 + rr) * N + colBase + c0;

  float y[32];
#pragma unroll
  for (int ch = 0; ch < 8; ++ch) {
#pragma unroll
    for (int j = 0; j < 4; ++j) y[ch * 4 + j] = srow[ch * 4 + j];
  }
  if (GELU) {
#pragma unroll
    for (int i = 0; i < 32; ++i) y[i] = gelu_f(y[i]);
  }
  if (RES) {
#pragma unroll
    for (int ch = 0; ch < 8; ++ch) {
      v4f rv = *(const v4f*)(resIn + gbase + ch * 4);
#pragma unroll
      for (int j = 0; j < 4; ++j) y[ch * 4 + j] += rv[j];
    }
  }
  if (OUTF32) {
#pragma unroll
    for (int ch = 0; ch < 8; ++ch) {
      v4f v;
#pragma unroll
      for (int j = 0; j < 4; ++j) v[j] = y[ch * 4 + j];
      *(v4f*)(outF + gbase + ch * 4) = v;
    }
  }
  if (OUTF16) {
#pragma unroll
    for (int ch = 0; ch < 4; ++ch) {
      v8h v;
#pragma unroll
      for (int j = 0; j < 8; ++j) v[j] = (h16)y[ch * 8 + j];
      *(v8h*)(outH + gbase + ch * 8) = v;
    }
  }
}

// --------------------------- embedding (WMMA) ------------------------------
// H[M,64] = BN(X[M,32] @ W[32,64] + b); 256 threads = 8 waves, 128 rows/block.
template <bool OUTH, bool OUTF>
__global__ void embed_kernel(const float* __restrict__ X, int M,
                             const h16* __restrict__ Wsw,  // swizzled 32x64
                             const float* __restrict__ b,
                             const float* __restrict__ rm,
                             const float* __restrict__ rv,
                             const float* __restrict__ g,
                             const float* __restrict__ be,
                             h16* __restrict__ outH,
                             float* __restrict__ outF) {
  __shared__ h16   Wsh[32 * 64];
  __shared__ float scale[64], shift[64];
  __shared__ float stage[8][16 * STAGE_LD];
  for (int i = threadIdx.x; i < 32 * 64 / 8; i += blockDim.x)
    ((uint4*)Wsh)[i] = ((const uint4*)Wsw)[i];
  if (threadIdx.x < 64) {
    int n = threadIdx.x;
    float inv = rsqrtf(rv[n] + 1e-5f);
    float sc = g[n] * inv;
    scale[n] = sc;
    shift[n] = be[n] - rm[n] * sc + b[n] * sc;
  }
  __syncthreads();

  int wave = threadIdx.x >> 5;
  int lane = threadIdx.x & 31;
  int row0 = (blockIdx.x * 8 + wave) * 16;
  if (row0 >= M) return;
  const float* Arow = X + (size_t)row0 * 32;
  __builtin_prefetch(Arow + (size_t)128 * 32, 0, 1);
  v16h a = load_a_frag_f(Arow, 32, 0);
  float* st = stage[wave];
#pragma unroll
  for (int t = 0; t < 4; ++t) {
    v8f c = {};
    c = wmma16x16x32(a, load_b_swz(Wsh, 4, 0, t), c);
    int n = t * 16 + (lane & 15);
    float sc = scale[n], sh = shift[n];
#pragma unroll
    for (int r = 0; r < 8; ++r)
      st[(r + 8 * (lane >> 4)) * STAGE_LD + t * 16 + (lane & 15)] = c[r] * sc + sh;
  }
  wave_flush<false, false, OUTH, OUTF>(st, nullptr, outH, outF, row0, 0, 64);
}

// ---------------------------- generic GEMM (WMMA) --------------------------
// C[M,N] = act(A[M,KD]h @ W[KD,N]h + bias) (+ residual) -> f16 and/or f32
// 256 threads = 8 waves, 128 rows/block; NT flushed in groups of 4 tiles.
template <int KD, int NT, bool GELU, bool RES, bool OUTF16, bool OUTF32>
__global__ void gemm_kernel(const h16* __restrict__ A,
                            const h16* __restrict__ Wsw,  // swizzled KDxN
                            const float* __restrict__ bias,
                            const float* __restrict__ resIn,
                            h16* __restrict__ outH,
                            float* __restrict__ outF, int M) {
  const int N = NT * 16;
  __shared__ h16   Wsh[KD * NT * 16];
  __shared__ float stage[8][16 * STAGE_LD];
  for (int i = threadIdx.x; i < KD * N / 8; i += blockDim.x)
    ((uint4*)Wsh)[i] = ((const uint4*)Wsw)[i];
  __syncthreads();

  int wave = threadIdx.x >> 5;
  int lane = threadIdx.x & 31;
  int row0 = (blockIdx.x * 8 + wave) * 16;
  if (row0 >= M) return;
  const h16* Arow = A + (size_t)row0 * KD;
  __builtin_prefetch(Arow + (size_t)128 * KD, 0, 1);

  v16h af[KD / 32];
#pragma unroll
  for (int ks = 0; ks < KD / 32; ++ks) af[ks] = load_a_frag_h(Arow, KD, ks * 32);

  float* st = stage[wave];
#pragma unroll
  for (int tg = 0; tg < NT / 4; ++tg) {
#pragma unroll
    for (int tl = 0; tl < 4; ++tl) {
      int t = tg * 4 + tl;
      v8f c = {};
#pragma unroll
      for (int ks = 0; ks < KD / 32; ++ks)
        c = wmma16x16x32(af[ks], load_b_swz(Wsh, NT, ks, t), c);
      float bcol = bias[t * 16 + (lane & 15)];
#pragma unroll
      for (int r = 0; r < 8; ++r)
        st[(r + 8 * (lane >> 4)) * STAGE_LD + tl * 16 + (lane & 15)] = c[r] + bcol;
    }
    wave_flush<GELU, RES, OUTF16, OUTF32>(st, resIn, outH, outF, row0, tg * 64, N);
  }
}

// -------------------- fused K/V projection + attention ---------------------
// Block = 2 anchors (128 neighbor rows), 128 threads (4 waves).
__global__ void attn_kernel(const h16* __restrict__ hnei,
                            const h16* __restrict__ Wk16,  // swizzled 64x64
                            const h16* __restrict__ Wv16,  // swizzled 64x64
                            const float* __restrict__ bk,
                            const float* __restrict__ bv,
                            const float* __restrict__ qbuf,   // [B,64] f32
                            const float* __restrict__ ew,     // [B,4,64]
                            const float* __restrict__ log_tau,
                            h16* __restrict__ ctxOut) {
  __shared__ h16   Wksh[64 * 64];
  __shared__ h16   Wvsh[64 * 64];
  __shared__ h16   kbuf[128 * 64];
  __shared__ h16   vbuf[128 * 64];
  __shared__ float logit[2 * 4 * 64];
  __shared__ float qsh[128];

  int a0 = blockIdx.x * 2;
  int wave = threadIdx.x >> 5;
  int lane = threadIdx.x & 31;

  for (int i = threadIdx.x; i < 64 * 64 / 8; i += blockDim.x) {
    ((uint4*)Wksh)[i] = ((const uint4*)Wk16)[i];
    ((uint4*)Wvsh)[i] = ((const uint4*)Wv16)[i];
  }
  qsh[threadIdx.x] = qbuf[(size_t)a0 * 64 + threadIdx.x];
  __syncthreads();

  // 1) K and V tiles: 2 mats x 8 row-tiles x 4 col-tiles = 64 jobs, 2 WMMA each
  for (int j = wave; j < 64; j += 4) {
    int mat = j & 1;
    int rt  = (j >> 1) & 7;
    int ct  = (j >> 4) & 3;
    const h16* Wm = mat ? Wvsh : Wksh;
    const float* bm = mat ? bv : bk;
    const h16* Arow = hnei + ((size_t)a0 * 64 + rt * 16) * 64;
    v8f c = {};
    c = wmma16x16x32(load_a_frag_h(Arow, 64, 0),  load_b_swz(Wm, 4, 0, ct), c);
    c = wmma16x16x32(load_a_frag_h(Arow, 64, 32), load_b_swz(Wm, 4, 1, ct), c);
    h16* dst = mat ? vbuf : kbuf;
    int n = ct * 16 + (lane & 15);
    float bcol = bm[n];
#pragma unroll
    for (int r = 0; r < 8; ++r) {
      int m = rt * 16 + r + 8 * (lane >> 4);
      dst[m * 64 + n] = (h16)(c[r] + bcol);
    }
  }
  __syncthreads();

  // 2) logits[a,h,n] = q.k / (sqrt(16)*tau[h]) + log(ew[a,view1,n] + 1e-9)
  for (int i = threadIdx.x; i < 512; i += 128) {
    int a = i >> 8;
    int h = (i >> 6) & 3;
    int n = i & 63;
    float acc = 0.0f;
#pragma unroll
    for (int d = 0; d < 16; ++d)
      acc += qsh[a * 64 + h * 16 + d] * (float)kbuf[(a * 64 + n) * 64 + h * 16 + d];
    float tau = __expf(log_tau[h]);
    float w = ew[((size_t)(a0 + a) * 4 + 1) * 64 + n];
    logit[i] = acc * (0.25f / tau) + __logf(w + 1e-9f);
  }
  __syncthreads();

  // 3) softmax over n per (a,h): 8 rows, 16 threads/row
  {
    int rrow = threadIdx.x >> 4;
    int l16  = threadIdx.x & 15;
    float* L = logit + rrow * 64;
    float mx = -1e30f;
    for (int n = l16; n < 64; n += 16) mx = fmaxf(mx, L[n]);
    for (int o = 8; o > 0; o >>= 1) mx = fmaxf(mx, __shfl_xor(mx, o, 16));
    float e[4];
    float sum = 0.0f;
    int ii = 0;
    for (int n = l16; n < 64; n += 16, ++ii) { e[ii] = __expf(L[n] - mx); sum += e[ii]; }
    for (int o = 8; o > 0; o >>= 1) sum += __shfl_xor(sum, o, 16);
    float invs = 1.0f / sum;
    ii = 0;
    for (int n = l16; n < 64; n += 16, ++ii) L[n] = e[ii] * invs;
  }
  __syncthreads();

  // 4) ctx[a,d] = sum_n attn[a,h(d),n] * v[a*64+n, d]
  {
    int a = threadIdx.x >> 6;
    int d = threadIdx.x & 63;
    int h = d >> 4;
    const float* Arow = logit + (a * 4 + h) * 64;
    float acc = 0.0f;
#pragma unroll 8
    for (int n = 0; n < 64; ++n)
      acc += Arow[n] * (float)vbuf[(a * 64 + n) * 64 + d];
    ctxOut[(size_t)(a0 + a) * 64 + d] = (h16)acc;
  }
}

// ------------------------------- LayerNorm ---------------------------------
__global__ void ln_kernel(const float* __restrict__ in,
                          float* __restrict__ outF, h16* __restrict__ outH,
                          const float* __restrict__ g,
                          const float* __restrict__ b, int M) {
  int row = blockIdx.x * 8 + (threadIdx.x >> 5);
  int lane = threadIdx.x & 31;
  if (row >= M) return;
  const float* x = in + (size_t)row * 64;
  float x0 = x[lane], x1 = x[lane + 32];
  float s = x0 + x1;
  for (int o = 16; o > 0; o >>= 1) s += __shfl_xor(s, o, 32);
  float mean = s * (1.0f / 64.0f);
  float d0 = x0 - mean, d1 = x1 - mean;
  float v = d0 * d0 + d1 * d1;
  for (int o = 16; o > 0; o >>= 1) v += __shfl_xor(v, o, 32);
  float inv = rsqrtf(v * (1.0f / 64.0f) + 1e-5f);
  float y0 = d0 * inv * g[lane] + b[lane];
  float y1 = d1 * inv * g[lane + 32] + b[lane + 32];
  outF[(size_t)row * 64 + lane] = y0;
  outF[(size_t)row * 64 + lane + 32] = y1;
  outH[(size_t)row * 64 + lane] = (h16)y0;
  outH[(size_t)row * 64 + lane + 32] = (h16)y1;
}

// --------------------------------- head ------------------------------------
__global__ void head_kernel(const float* __restrict__ h,
                            const float* __restrict__ g,
                            const float* __restrict__ b,
                            const h16* __restrict__ W1,   // 64x32 plain
                            const float* __restrict__ c1,
                            const h16* __restrict__ W2,   // 32x8 plain
                            const float* __restrict__ c2,
                            float* __restrict__ pred, int M) {
  __shared__ float hn[8][64];
  __shared__ float t[8][32];
  int wl = threadIdx.x >> 5;
  int lane = threadIdx.x & 31;
  int row = blockIdx.x * 8 + wl;
  if (row >= M) return;
  const float* x = h + (size_t)row * 64;
  float x0 = x[lane], x1 = x[lane + 32];
  float s = x0 + x1;
  for (int o = 16; o > 0; o >>= 1) s += __shfl_xor(s, o, 32);
  float mean = s * (1.0f / 64.0f);
  float d0 = x0 - mean, d1 = x1 - mean;
  float v = d0 * d0 + d1 * d1;
  for (int o = 16; o > 0; o >>= 1) v += __shfl_xor(v, o, 32);
  float inv = rsqrtf(v * (1.0f / 64.0f) + 1e-5f);
  hn[wl][lane]      = d0 * inv * g[lane] + b[lane];
  hn[wl][lane + 32] = d1 * inv * g[lane + 32] + b[lane + 32];
  __syncthreads();
  float acc = c1[lane];
#pragma unroll 8
  for (int k = 0; k < 64; ++k) acc += hn[wl][k] * (float)W1[k * 32 + lane];
  t[wl][lane] = gelu_f(acc);
  __syncthreads();
  if (lane < 8) {
    float a2 = c2[lane];
#pragma unroll 8
    for (int k = 0; k < 32; ++k) a2 += t[wl][k] * (float)W2[k * 8 + lane];
    pred[(size_t)row * 8 + lane] = a2;
  }
}

// ------------------------------- launcher ----------------------------------
// Input order assumption (setup_inputs dict order; params flattened with
// sorted dict keys, lists in order — jax pytree convention):
//  0 x_anc  1 g_anc  2 x_nei  3 ew_anc
//  4..9   emb:  W, b, be, g, rm, rv
//  10..15 head: W1, W2, b, c1, c2, g
//  16+16*l : layer l: W1,W2,Wk,Wo,Wq,Wv,b1,b2,bk,bo,bq,bv,c1,c2,g1,g2
//  48 log_tau

extern "C" void kernel_launch(void* const* d_in, const int* in_sizes, int n_in,
                              void* d_out, int out_size, void* d_ws, size_t ws_size,
                              hipStream_t stream) {
  const float* x_anc = (const float*)d_in[0];
  const float* x_nei = (const float*)d_in[2];
  const float* ew    = (const float*)d_in[3];
  const float* embW  = (const float*)d_in[4];
  const float* embB  = (const float*)d_in[5];
  const float* embBe = (const float*)d_in[6];
  const float* embG  = (const float*)d_in[7];
  const float* embRm = (const float*)d_in[8];
  const float* embRv = (const float*)d_in[9];
  const float* hW1   = (const float*)d_in[10];
  const float* hW2   = (const float*)d_in[11];
  const float* hB    = (const float*)d_in[12];
  const float* hC1   = (const float*)d_in[13];
  const float* hC2   = (const float*)d_in[14];
  const float* hG    = (const float*)d_in[15];
  const float* log_tau = (const float*)d_in[48];

  struct Layer {
    const float *W1, *W2, *Wk, *Wo, *Wq, *Wv;
    const float *b1, *b2, *bk, *bo, *bq, *bv, *c1, *c2, *g1, *g2;
  } L[2];
  for (int l = 0; l < 2; ++l) {
    int o = 16 + 16 * l;
    L[l].W1 = (const float*)d_in[o + 0];  L[l].W2 = (const float*)d_in[o + 1];
    L[l].Wk = (const float*)d_in[o + 2];  L[l].Wo = (const float*)d_in[o + 3];
    L[l].Wq = (const float*)d_in[o + 4];  L[l].Wv = (const float*)d_in[o + 5];
    L[l].b1 = (const float*)d_in[o + 6];  L[l].b2 = (const float*)d_in[o + 7];
    L[l].bk = (const float*)d_in[o + 8];  L[l].bo = (const float*)d_in[o + 9];
    L[l].bq = (const float*)d_in[o + 10]; L[l].bv = (const float*)d_in[o + 11];
    L[l].c1 = (const float*)d_in[o + 12]; L[l].c2 = (const float*)d_in[o + 13];
    L[l].g1 = (const float*)d_in[o + 14]; L[l].g2 = (const float*)d_in[o + 15];
  }

  // ---- workspace layout ----
  char* ws = (char*)d_ws;
  const size_t MN = (size_t)NBATCH * KNEI;          // 524288 neighbor rows
  size_t off = 0;
  h16*   hnei16 = (h16*)(ws + off);   off += MN * 64 * 2;           // 64 MB
  float* hanc32 = (float*)(ws + off); off += (size_t)NBATCH * 64 * 4;
  h16*   hanc16 = (h16*)(ws + off);   off += (size_t)NBATCH * 64 * 2;
  float* hln32  = (float*)(ws + off); off += (size_t)NBATCH * 64 * 4;
  h16*   hln16  = (h16*)(ws + off);   off += (size_t)NBATCH * 64 * 2;
  float* tmp32  = (float*)(ws + off); off += (size_t)NBATCH * 64 * 4;
  float* q32    = (float*)(ws + off); off += (size_t)NBATCH * 64 * 4;
  h16*   ctx16  = (h16*)(ws + off);   off += (size_t)NBATCH * 64 * 2;
  h16*   ff16   = (h16*)(ws + off);   off += (size_t)NBATCH * 128 * 2;
  h16*   wpool  = (h16*)(ws + off);

  h16* embW16 = wpool;                       // 2048 (swizzled)
  h16* lw[2][6];                             // Wq,Wk,Wv,Wo,W1,W2 (swizzled)
  size_t wo = 2048;
  for (int l = 0; l < 2; ++l) {
    lw[l][0] = wpool + wo; wo += 4096;       // Wq
    lw[l][1] = wpool + wo; wo += 4096;       // Wk
    lw[l][2] = wpool + wo; wo += 4096;       // Wv
    lw[l][3] = wpool + wo; wo += 4096;       // Wo
    lw[l][4] = wpool + wo; wo += 8192;       // W1
    lw[l][5] = wpool + wo; wo += 8192;       // W2
  }
  h16* hW1_16 = wpool + wo; wo += 2048;      // plain f16
  h16* hW2_16 = wpool + wo; wo += 256;       // plain f16

  // ---- pack weights (swizzled for WMMA GEMMs, plain for the tiny head) ----
  struct SJob { const float* s; h16* d; int kd, n; } sjobs[] = {
    {embW, embW16, 32, 64},
    {L[0].Wq, lw[0][0], 64, 64}, {L[0].Wk, lw[0][1], 64, 64},
    {L[0].Wv, lw[0][2], 64, 64}, {L[0].Wo, lw[0][3], 64, 64},
    {L[0].W1, lw[0][4], 64, 128}, {L[0].W2, lw[0][5], 128, 64},
    {L[1].Wq, lw[1][0], 64, 64}, {L[1].Wk, lw[1][1], 64, 64},
    {L[1].Wv, lw[1][2], 64, 64}, {L[1].Wo, lw[1][3], 64, 64},
    {L[1].W1, lw[1][4], 64, 128}, {L[1].W2, lw[1][5], 128, 64},
  };
  for (auto& j : sjobs)
    pack_swz_kernel<<<(j.kd * j.n + 255) / 256, 256, 0, stream>>>(j.s, j.d, j.kd, j.n);
  pack_f16_kernel<<<8, 256, 0, stream>>>(hW1, hW1_16, 2048);
  pack_f16_kernel<<<1, 256, 0, stream>>>(hW2, hW2_16, 256);

  // ---- embedding (anchors write f16+f32 master, neighbors f16 only) ----
  embed_kernel<true, true><<<NBATCH / 128, 256, 0, stream>>>(
      x_anc, NBATCH, embW16, embB, embRm, embRv, embG, embBe, hanc16, hanc32);
  embed_kernel<true, false><<<(int)(MN / 128), 256, 0, stream>>>(
      x_nei, (int)MN, embW16, embB, embRm, embRv, embG, embBe, hnei16, nullptr);

  // ---- transformer layers ----
  for (int l = 0; l < 2; ++l) {
    gemm_kernel<64, 4, false, false, false, true><<<NBATCH / 128, 256, 0, stream>>>(
        hanc16, lw[l][0], L[l].bq, nullptr, nullptr, q32, NBATCH);
    attn_kernel<<<NBATCH / 2, 128, 0, stream>>>(
        hnei16, lw[l][1], lw[l][2], L[l].bk, L[l].bv, q32, ew, log_tau, ctx16);
    gemm_kernel<64, 4, false, true, false, true><<<NBATCH / 128, 256, 0, stream>>>(
        ctx16, lw[l][3], L[l].bo, hanc32, nullptr, tmp32, NBATCH);
    ln_kernel<<<NBATCH / 8, 256, 0, stream>>>(tmp32, hln32, hln16, L[l].g1, L[l].b1, NBATCH);
    gemm_kernel<64, 8, true, false, true, false><<<NBATCH / 128, 256, 0, stream>>>(
        hln16, lw[l][4], L[l].c1, nullptr, ff16, nullptr, NBATCH);
    gemm_kernel<128, 4, false, true, false, true><<<NBATCH / 128, 256, 0, stream>>>(
        ff16, lw[l][5], L[l].c2, hln32, nullptr, tmp32, NBATCH);
    ln_kernel<<<NBATCH / 8, 256, 0, stream>>>(tmp32, hanc32, hanc16, L[l].g2, L[l].b2, NBATCH);
  }

  // ---- head + pi/tau ----
  float* out = (float*)d_out;
  head_kernel<<<NBATCH / 8, 256, 0, stream>>>(
      hanc32, hG, hB, hW1_16, hC1, hW2_16, hC2, out, NBATCH);
  fill_pi_tau_kernel<<<(NBATCH * 16 + 255) / 256, 256, 0, stream>>>(out, log_tau);
}
